// GATEncoder_6983616823541
// MI455X (gfx1250) — compile-verified, matrix operand order checked by reading, not run on hardware
//
#include <hip/hip_runtime.h>
#include <hip/hip_bf16.h>
#include <math.h>

#define NEG_SLOPE 0.2f
#define BN_EPS 1e-5f

typedef __attribute__((ext_vector_type(16))) __bf16 v16bf;
typedef __attribute__((ext_vector_type(8)))  float  v8f;

struct B32x8 { uint4 a; uint4 b; };   // 32 bytes == v16bf

// ---------------------------------------------------------------- utilities
__global__ void fill_f32(float* p, float v, long n) {
    long i = (long)blockIdx.x * blockDim.x + threadIdx.x;
    if (i < n) p[i] = v;
}

__global__ void f32_to_bf16_k(const float* __restrict__ in, __bf16* __restrict__ out, long n) {
    long i = (long)blockIdx.x * blockDim.x + threadIdx.x;
    if (i < n) out[i] = (__bf16)in[i];
}

// W: K x Nout fp32 (row-major)  ->  Wt: Nout x K bf16 (row-major)
__global__ void transpose_to_bf16_k(const float* __restrict__ W, __bf16* __restrict__ Wt,
                                    int K, int Nout) {
    int i = blockIdx.x * blockDim.x + threadIdx.x;
    if (i < K * Nout) {
        int k = i / Nout, n = i % Nout;
        Wt[(long)n * K + k] = (__bf16)W[i];
    }
}

// ------------------------------------------------------------- WMMA GEMM
// Loads one lane's half-row chunk for the 16-bit WMMA A/B layout:
//   element pairs p=0..3 -> k = kk*32 + grp*8 + 2p   (contiguous 16B)
//   element pairs p=4..7 -> k = kk*32 + 16 + grp*8 + 2(p-4) (contiguous 16B)
__device__ __forceinline__ v16bf load_wmma_row(const __bf16* row, int kk, int grp) {
    B32x8 raw;
    raw.a = *(const uint4*)(row + kk * 32 + grp * 8);
    raw.b = *(const uint4*)(row + kk * 32 + 16 + grp * 8);
    return __builtin_bit_cast(v16bf, raw);
}

// C(M x Nout) = A(M x K, bf16) * Bt^T (Bt is Nout x K, bf16) + bias
template<int KD>
__global__ void wmma_gemm_bias(const __bf16* __restrict__ A, const __bf16* __restrict__ Bt,
                               const float* __restrict__ bias, float* __restrict__ C,
                               int M, int Nout) {
    const int lane = threadIdx.x & 31;
    const int wave = (blockIdx.x * blockDim.x + threadIdx.x) >> 5;
    const int row0 = wave << 4;
    if (row0 >= M) return;                 // wave-uniform: EXEC stays all-ones for WMMA
    const int mrow = row0 + (lane & 15);
    const int grp  = lane >> 4;
    const int ncol = lane & 15;

    // cache this wave's 16 x KD A-strip in registers (KD/32 WMMA operands)
    v16bf aReg[KD / 32];
    const __bf16* Arow = A + (long)mrow * KD;
#pragma unroll
    for (int kk = 0; kk < KD / 32; ++kk) aReg[kk] = load_wmma_row(Arow, kk, grp);

    for (int n0 = 0; n0 < Nout; n0 += 16) {
        v8f acc = {};
        const __bf16* Brow = Bt + (long)(n0 + ncol) * KD;
#pragma unroll
        for (int kk = 0; kk < KD / 32; ++kk) {
            v16bf b = load_wmma_row(Brow, kk, grp);
            acc = __builtin_amdgcn_wmma_f32_16x16x32_bf16(
                      false, aReg[kk], false, b, (short)0, acc, false, false);
        }
        float bb = bias[n0 + ncol];
#pragma unroll
        for (int r = 0; r < 8; ++r)       // C layout: VGPR r -> row grp*8+r, lane -> col
            C[(long)(row0 + grp * 8 + r) * Nout + (n0 + ncol)] = acc[r] + bb;
    }
}

// ------------------------------------------------------------- edge phase
__device__ __forceinline__ void atomicMaxF(float* addr, float val) {
    if (val >= 0.0f) atomicMax((int*)addr, __float_as_int(val));
    else             atomicMin((unsigned int*)addr, __float_as_uint(val));
}

__device__ __forceinline__ void edge_endpoints(int e, int E, const int* __restrict__ ei,
                                               int& s, int& d) {
    if (e < E) { s = ei[e]; d = ei[E + e]; }
    else       { s = d = e - E; }          // self loops appended
}

// logit[e][h] = att[h] . leaky_relu(xl[s] + xr[d]);  mmax[d][h] = segment max
__global__ void edge_score_max(const float* __restrict__ xl, const float* __restrict__ xr,
                               const int* __restrict__ ei, const float* __restrict__ att,
                               float* __restrict__ logit, float* __restrict__ mmax,
                               int E, int Nn, int C) {
    int e = blockIdx.x * blockDim.x + threadIdx.x;
    if (e >= E + Nn) return;
    int s, d; edge_endpoints(e, E, ei, s, d);
    const int D = 2 * C;
    const float4* pl = (const float4*)(xl + (long)s * D);
    const float4* pr = (const float4*)(xr + (long)d * D);
#pragma unroll
    for (int h = 0; h < 2; ++h) {
        const float4* pa = (const float4*)(att + h * C);
        float acc = 0.0f;
        for (int c4 = 0; c4 < C / 4; ++c4) {
            float4 va = pl[h * (C / 4) + c4];
            float4 vb = pr[h * (C / 4) + c4];
            float4 wa = pa[c4];
            float t;
            t = va.x + vb.x; acc += wa.x * (t > 0.0f ? t : NEG_SLOPE * t);
            t = va.y + vb.y; acc += wa.y * (t > 0.0f ? t : NEG_SLOPE * t);
            t = va.z + vb.z; acc += wa.z * (t > 0.0f ? t : NEG_SLOPE * t);
            t = va.w + vb.w; acc += wa.w * (t > 0.0f ? t : NEG_SLOPE * t);
        }
        logit[(long)e * 2 + h] = acc;
        atomicMaxF(&mmax[(long)d * 2 + h], acc);
    }
}

// ex = exp(logit - max[dst]); den[dst] += ex  (logit buffer overwritten with ex)
__global__ void edge_expsum(const int* __restrict__ ei, float* __restrict__ logit,
                            const float* __restrict__ mmax, float* __restrict__ den,
                            int E, int Nn) {
    int e = blockIdx.x * blockDim.x + threadIdx.x;
    if (e >= E + Nn) return;
    int s, d; edge_endpoints(e, E, ei, s, d); (void)s;
#pragma unroll
    for (int h = 0; h < 2; ++h) {
        float ex = __expf(logit[(long)e * 2 + h] - mmax[(long)d * 2 + h]);
        logit[(long)e * 2 + h] = ex;
        atomicAdd(&den[(long)d * 2 + h], ex);
    }
}

// agg[dst] += alpha * xl[src]
__global__ void edge_aggregate(const float* __restrict__ xl, const int* __restrict__ ei,
                               const float* __restrict__ ex, const float* __restrict__ den,
                               float* __restrict__ agg, int E, int Nn, int C) {
    int e = blockIdx.x * blockDim.x + threadIdx.x;
    if (e >= E + Nn) return;
    int s, d; edge_endpoints(e, E, ei, s, d);
    const int D = 2 * C;
#pragma unroll
    for (int h = 0; h < 2; ++h) {
        float alpha = ex[(long)e * 2 + h] / den[(long)d * 2 + h];
        const float4* pl = (const float4*)(xl + (long)s * D + h * C);
        float* pd = agg + (long)d * D + h * C;
        for (int c4 = 0; c4 < C / 4; ++c4) {
            float4 v = pl[c4];
            atomicAdd(pd + 4 * c4 + 0, v.x * alpha);
            atomicAdd(pd + 4 * c4 + 1, v.y * alpha);
            atomicAdd(pd + 4 * c4 + 2, v.z * alpha);
            atomicAdd(pd + 4 * c4 + 3, v.w * alpha);
        }
    }
}

// ------------------------------------------------------------- batch norm
__global__ void bn_stats(const float* __restrict__ X, float* __restrict__ mean,
                         float* __restrict__ rstd, int M, int D) {
    int ch = blockIdx.x;
    float s = 0.0f, s2 = 0.0f;
    for (int i = threadIdx.x; i < M; i += blockDim.x) {
        float v = X[(long)i * D + ch];
        s += v; s2 += v * v;
    }
    __shared__ float sh1[256], sh2[256];
    int t = threadIdx.x;
    sh1[t] = s; sh2[t] = s2;
    __syncthreads();
    for (int o = 128; o > 0; o >>= 1) {
        if (t < o) { sh1[t] += sh1[t + o]; sh2[t] += sh2[t + o]; }
        __syncthreads();
    }
    if (t == 0) {
        float mu  = sh1[0] / (float)M;
        float var = sh2[0] / (float)M - mu * mu;
        mean[ch] = mu;
        rstd[ch] = rsqrtf(var + BN_EPS);
    }
}

__global__ void bn_relu_bf16(const float* __restrict__ X, const float* __restrict__ mean,
                             const float* __restrict__ rstd, const float* __restrict__ gamma,
                             const float* __restrict__ beta, __bf16* __restrict__ out,
                             long n, int D) {
    long i = (long)blockIdx.x * blockDim.x + threadIdx.x;
    if (i >= n) return;
    int ch = (int)(i % D);
    float v = (X[i] - mean[ch]) * rstd[ch] * gamma[ch] + beta[ch];
    out[i] = (__bf16)(v > 0.0f ? v : 0.0f);
}

// o[i][c] = mean over heads of agg2 + bias2
__global__ void head_mean_bias(const float* __restrict__ agg, const float* __restrict__ bias,
                               float* __restrict__ o, int Nn, int C) {
    long i = (long)blockIdx.x * blockDim.x + threadIdx.x;
    if (i >= (long)Nn * C) return;
    long r = i / C; int c = (int)(i % C);
    o[i] = 0.5f * (agg[r * 2 * C + c] + agg[r * 2 * C + C + c]) + bias[c];
}

// out = [ (o[:, :OUTC]-mu)*rstd , softplus(o[:, OUTC:]) ]
__global__ void final_out(const float* __restrict__ o, const float* __restrict__ mean,
                          const float* __restrict__ rstd, float* __restrict__ out,
                          int Nn, int C, int OUTC) {
    long i = (long)blockIdx.x * blockDim.x + threadIdx.x;
    if (i >= (long)Nn * C) return;
    int c = (int)(i % C);
    float v = o[i];
    if (c < OUTC) out[i] = (v - mean[c]) * rstd[c];
    else          out[i] = (v > 20.0f) ? v : log1pf(__expf(v));
}

// ---------------------------------------------------------------- launch
extern "C" void kernel_launch(void* const* d_in, const int* in_sizes, int n_in,
                              void* d_out, int out_size, void* d_ws, size_t ws_size,
                              hipStream_t stream) {
    (void)n_in; (void)out_size; (void)ws_size;
    const float* x      = (const float*)d_in[0];
    const int*   ei     = (const int*)  d_in[1];
    const float* W_l1   = (const float*)d_in[2];
    const float* b_l1   = (const float*)d_in[3];
    const float* W_r1   = (const float*)d_in[4];
    const float* b_r1   = (const float*)d_in[5];
    const float* att1   = (const float*)d_in[6];
    /* d_in[7] = bias1 : per-channel constant cancels exactly inside BatchNorm */
    const float* gamma1 = (const float*)d_in[8];
    const float* beta1  = (const float*)d_in[9];
    const float* W_l2   = (const float*)d_in[10];
    const float* b_l2   = (const float*)d_in[11];
    const float* W_r2   = (const float*)d_in[12];
    const float* b_r2   = (const float*)d_in[13];
    const float* att2   = (const float*)d_in[14];
    const float* bias2  = (const float*)d_in[15];

    const int IN = 128, D1 = 128, C1 = 64, C2 = 128, D2 = 256, OUTC = 64;
    const int Nn = in_sizes[0] / IN;
    const int E  = in_sizes[1] / 2;
    const int Et = E + Nn;

    // workspace carving (256B aligned)
    size_t off = 0;
    auto alloc = [&](size_t bytes) -> void* {
        void* p = (char*)d_ws + off;
        off += (bytes + 255) & ~(size_t)255;
        return p;
    };
    __bf16* x_bf  = (__bf16*)alloc((size_t)Nn * IN * sizeof(__bf16));
    __bf16* Wt1l  = (__bf16*)alloc((size_t)IN * D1 * sizeof(__bf16));
    __bf16* Wt1r  = (__bf16*)alloc((size_t)IN * D1 * sizeof(__bf16));
    __bf16* Wt2l  = (__bf16*)alloc((size_t)D1 * D2 * sizeof(__bf16));
    __bf16* Wt2r  = (__bf16*)alloc((size_t)D1 * D2 * sizeof(__bf16));
    float*  xl1   = (float*) alloc((size_t)Nn * D1 * sizeof(float));
    float*  xr1   = (float*) alloc((size_t)Nn * D1 * sizeof(float));
    float*  xl2   = (float*) alloc((size_t)Nn * D2 * sizeof(float));
    float*  xr2   = (float*) alloc((size_t)Nn * D2 * sizeof(float));
    float*  logit = (float*) alloc((size_t)Et * 2 * sizeof(float));
    float*  mmax  = (float*) alloc((size_t)Nn * 2 * sizeof(float));
    float*  den   = (float*) alloc((size_t)Nn * 2 * sizeof(float));
    float*  agg1  = (float*) alloc((size_t)Nn * D1 * sizeof(float));
    __bf16* h_bf  = (__bf16*)alloc((size_t)Nn * D1 * sizeof(__bf16));
    float*  agg2  = (float*) alloc((size_t)Nn * D2 * sizeof(float));
    float*  obuf  = (float*) alloc((size_t)Nn * C2 * sizeof(float));
    float*  meanb = (float*) alloc((size_t)D1 * sizeof(float));
    float*  rstdb = (float*) alloc((size_t)D1 * sizeof(float));

    const int TB = 256;
    auto cdiv = [](long a, long b) { return (int)((a + b - 1) / b); };

    // 0) precision conversion + weight transpose (bf16 for WMMA B-operand layout)
    f32_to_bf16_k<<<cdiv((long)Nn * IN, TB), TB, 0, stream>>>(x, x_bf, (long)Nn * IN);
    transpose_to_bf16_k<<<cdiv(IN * D1, TB), TB, 0, stream>>>(W_l1, Wt1l, IN, D1);
    transpose_to_bf16_k<<<cdiv(IN * D1, TB), TB, 0, stream>>>(W_r1, Wt1r, IN, D1);
    transpose_to_bf16_k<<<cdiv(D1 * D2, TB), TB, 0, stream>>>(W_l2, Wt2l, D1, D2);
    transpose_to_bf16_k<<<cdiv(D1 * D2, TB), TB, 0, stream>>>(W_r2, Wt2r, D1, D2);

    const int rowTiles   = cdiv(Nn, 16);
    const int gemmBlocks = cdiv((long)rowTiles * 32, TB);

    // 1) layer-1 feature transforms (bf16 WMMA, fp32 accum)
    wmma_gemm_bias<128><<<gemmBlocks, TB, 0, stream>>>(x_bf, Wt1l, b_l1, xl1, Nn, D1);
    wmma_gemm_bias<128><<<gemmBlocks, TB, 0, stream>>>(x_bf, Wt1r, b_r1, xr1, Nn, D1);

    // 2) layer-1 edge softmax + aggregation (L2-resident gathers/scatters)
    const int eb = cdiv(Et, TB);
    fill_f32<<<cdiv((long)Nn * 2, TB), TB, 0, stream>>>(mmax, -3.0e38f, (long)Nn * 2);
    fill_f32<<<cdiv((long)Nn * 2, TB), TB, 0, stream>>>(den, 0.0f, (long)Nn * 2);
    fill_f32<<<cdiv((long)Nn * D1, TB), TB, 0, stream>>>(agg1, 0.0f, (long)Nn * D1);
    edge_score_max<<<eb, TB, 0, stream>>>(xl1, xr1, ei, att1, logit, mmax, E, Nn, C1);
    edge_expsum  <<<eb, TB, 0, stream>>>(ei, logit, mmax, den, E, Nn);
    edge_aggregate<<<eb, TB, 0, stream>>>(xl1, ei, logit, den, agg1, E, Nn, C1);

    // 3) BN1 + ReLU, emit bf16 activations for next GEMM
    bn_stats<<<D1, 256, 0, stream>>>(agg1, meanb, rstdb, Nn, D1);
    bn_relu_bf16<<<cdiv((long)Nn * D1, TB), TB, 0, stream>>>(agg1, meanb, rstdb,
                                                             gamma1, beta1, h_bf,
                                                             (long)Nn * D1, D1);

    // 4) layer-2 feature transforms
    wmma_gemm_bias<128><<<gemmBlocks, TB, 0, stream>>>(h_bf, Wt2l, b_l2, xl2, Nn, D2);
    wmma_gemm_bias<128><<<gemmBlocks, TB, 0, stream>>>(h_bf, Wt2r, b_r2, xr2, Nn, D2);

    // 5) layer-2 edge softmax + aggregation
    fill_f32<<<cdiv((long)Nn * 2, TB), TB, 0, stream>>>(mmax, -3.0e38f, (long)Nn * 2);
    fill_f32<<<cdiv((long)Nn * 2, TB), TB, 0, stream>>>(den, 0.0f, (long)Nn * 2);
    fill_f32<<<cdiv((long)Nn * D2, TB), TB, 0, stream>>>(agg2, 0.0f, (long)Nn * D2);
    edge_score_max<<<eb, TB, 0, stream>>>(xl2, xr2, ei, att2, logit, mmax, E, Nn, C2);
    edge_expsum  <<<eb, TB, 0, stream>>>(ei, logit, mmax, den, E, Nn);
    edge_aggregate<<<eb, TB, 0, stream>>>(xl2, ei, logit, den, agg2, E, Nn, C2);

    // 6) head mean + bias2, BN (affine-free, first 64 ch) + softplus tail
    head_mean_bias<<<cdiv((long)Nn * C2, TB), TB, 0, stream>>>(agg2, bias2, obuf, Nn, C2);
    bn_stats<<<OUTC, 256, 0, stream>>>(obuf, meanb, rstdb, Nn, C2);
    final_out<<<cdiv((long)Nn * C2, TB), TB, 0, stream>>>(obuf, meanb, rstdb,
                                                          (float*)d_out, Nn, C2, OUTC);
}
